// MultiQueryAttention_48412871360713
// MI455X (gfx1250) — matrix operand order repaired and checked by reading, not compiled, not run on hardware
//
#include <hip/hip_runtime.h>
#include <hip/hip_bf16.h>

typedef __attribute__((ext_vector_type(16))) _Float16 v16h;
typedef __attribute__((ext_vector_type(8)))  _Float16 v8h;
typedef __attribute__((ext_vector_type(8)))  float    v8f;

#define DIM_C   1024
#define NHEADS  16
#define HDIM    64
#define SEQ_T   2048

__device__ __forceinline__ float red_max16(float v) {
    v = fmaxf(v, __shfl_xor(v, 1, 32));
    v = fmaxf(v, __shfl_xor(v, 2, 32));
    v = fmaxf(v, __shfl_xor(v, 4, 32));
    v = fmaxf(v, __shfl_xor(v, 8, 32));
    return v;
}
__device__ __forceinline__ float red_sum16(float v) {
    v += __shfl_xor(v, 1, 32);
    v += __shfl_xor(v, 2, 32);
    v += __shfl_xor(v, 4, 32);
    v += __shfl_xor(v, 8, 32);
    return v;
}

// Build a 16-f16 fragment from two contiguous 8-f16 runs (2 x b128 loads).
__device__ __forceinline__ v16h frag2(const _Float16* p0, const _Float16* p1) {
    const v8h lo = *(const v8h*)p0;
    const v8h hi = *(const v8h*)p1;
    return __builtin_shufflevector(lo, hi, 0, 1, 2, 3, 4, 5, 6, 7,
                                           8, 9, 10, 11, 12, 13, 14, 15);
}

// ---------------------------------------------------------------------------
// GEMM: Out(M,N) = A(M,K)f16 @ Bt(N,K)f16^T + bias, f32 out.  K compile-time
// so per-nt strides fold into VMEM immediate offsets.  Ping-pong K loop
// (64 deep): stage loads refill one buffer set while WMMAs consume the other
// -> no register rotation copies, partial loadcnt waits.
// ---------------------------------------------------------------------------
template<int K>
__global__ __launch_bounds__(128)
void gemm_wmma(const _Float16* __restrict__ A, const _Float16* __restrict__ Bt,
               const float* __restrict__ bias, float* __restrict__ Out, int N)
{
    const int wave = threadIdx.x >> 5;
    const int lane = threadIdx.x & 31;
    const int half = lane >> 4;
    const int lr   = lane & 15;
    const int m0 = (blockIdx.y * 4 + wave) * 16;
    const int n0 = blockIdx.x * 64;
    const int kb  = half * 16;
    const int kbB = half * 16;
    constexpr int BS = 16 * K;                 // per-nt B stride (immediate)

    const _Float16* pA = A  + (size_t)(m0 + lr) * K + kb;
    const _Float16* pB = Bt + (size_t)(n0 + lr) * K + kbB;

    v8f acc[4] = {{}, {}, {}, {}};

    v16h a0 = frag2(pA, pA + 8);
    v16h b0[4];
    #pragma unroll
    for (int nt = 0; nt < 4; ++nt)
        b0[nt] = frag2(pB + nt * BS, pB + nt * BS + 8);

    constexpr int ITER = K / 64;
    for (int it = 0; it < ITER - 1; ++it) {
        // odd stage loads (k+32)
        const v16h a1 = frag2(pA + 32, pA + 40);
        v16h b1[4];
        #pragma unroll
        for (int nt = 0; nt < 4; ++nt)
            b1[nt] = frag2(pB + nt * BS + 32, pB + nt * BS + 40);
        if (it < ITER - 2) {                   // in-bounds prefetch, 1 it ahead
            __builtin_prefetch((const void*)(pA + 128), 0, 1);
            #pragma unroll
            for (int nt = 0; nt < 4; ++nt)
                __builtin_prefetch((const void*)(pB + nt * BS + 128), 0, 1);
        }
        // consume even stage
        #pragma unroll
        for (int nt = 0; nt < 4; ++nt)
            acc[nt] = __builtin_amdgcn_wmma_f32_16x16x32_f16(
                false, a0, false, b0[nt], (short)0, acc[nt], false, false);
        pA += 64;
        pB += 64;
        // refill even stage for next iteration (k+64)
        a0 = frag2(pA, pA + 8);
        #pragma unroll
        for (int nt = 0; nt < 4; ++nt)
            b0[nt] = frag2(pB + nt * BS, pB + nt * BS + 8);
        // consume odd stage
        #pragma unroll
        for (int nt = 0; nt < 4; ++nt)
            acc[nt] = __builtin_amdgcn_wmma_f32_16x16x32_f16(
                false, a1, false, b1[nt], (short)0, acc[nt], false, false);
    }
    {   // tail: stages K-64 (in a0/b0) and K-32
        const v16h a1 = frag2(pA + 32, pA + 40);
        v16h b1[4];
        #pragma unroll
        for (int nt = 0; nt < 4; ++nt)
            b1[nt] = frag2(pB + nt * BS + 32, pB + nt * BS + 40);
        #pragma unroll
        for (int nt = 0; nt < 4; ++nt)
            acc[nt] = __builtin_amdgcn_wmma_f32_16x16x32_f16(
                false, a0, false, b0[nt], (short)0, acc[nt], false, false);
        #pragma unroll
        for (int nt = 0; nt < 4; ++nt)
            acc[nt] = __builtin_amdgcn_wmma_f32_16x16x32_f16(
                false, a1, false, b1[nt], (short)0, acc[nt], false, false);
    }

    #pragma unroll
    for (int nt = 0; nt < 4; ++nt) {
        const int col = n0 + nt * 16 + lr;
        const float bv = bias[col];
        #pragma unroll
        for (int r = 0; r < 8; ++r) {
            const int row = m0 + r + 8 * half;            // C/D layout
            Out[(size_t)row * N + col] = acc[nt][r] + bv;
        }
    }
}

// ---------------------------------------------------------------------------
// Prep kernels
// ---------------------------------------------------------------------------
__global__ void cvt_kernel(const float* __restrict__ In,
                           _Float16* __restrict__ Out, int n)
{
    const int i = blockIdx.x * blockDim.x + threadIdx.x;
    if (i < n) Out[i] = (_Float16)In[i];
}

__global__ void wtrans_kernel(const float* __restrict__ W,
                              _Float16* __restrict__ Wt, int K, int N)
{
    const int idx = blockIdx.x * blockDim.x + threadIdx.x;
    if (idx >= K * N) return;
    const int k = idx % K;
    const int n = idx / K;
    Wt[idx] = (_Float16)W[(size_t)k * N + n];
}

__global__ void vtrans_kernel(const float* __restrict__ V,
                              _Float16* __restrict__ Vt, int B)
{
    const int idx = blockIdx.x * blockDim.x + threadIdx.x;
    if (idx >= B * SEQ_T * HDIM) return;
    const int t = idx % SEQ_T;
    const int d = (idx / SEQ_T) % HDIM;
    const int b = idx / (SEQ_T * HDIM);
    Vt[idx] = (_Float16)V[((size_t)b * SEQ_T + t) * HDIM + d];
}

__global__ void rope_kernel(const float* __restrict__ In,
                            _Float16* __restrict__ Out,
                            int B, int T, int H)
{
    const int idx = blockIdx.x * blockDim.x + threadIdx.x;
    if (idx >= B * T * H * 32) return;
    const int j  = idx & 15;
    const int hs = (idx >> 4) & 1;
    const int h  = (idx >> 5) % H;
    const int tt = (idx >> 5) / H;
    const int t  = tt % T;
    const int b  = tt / T;

    const float* src = In + (((size_t)(b * T + t)) * H + h) * HDIM + hs * 32;
    const float e = src[2 * j];
    const float o = src[2 * j + 1];
    const float theta = __expf(-(float)(2 * j) * (1.0f / 32.0f) * 9.210340372f);
    const float fr = (float)t * theta;
    const float s = __sinf(fr);
    const float c = __cosf(fr);

    _Float16* dst = Out + (((size_t)(b * H + h)) * T + t) * HDIM + hs * 32;
    dst[j]      = (_Float16)(e * c - o * s);
    dst[16 + j] = (_Float16)(e * s + o * c);
}

// ---------------------------------------------------------------------------
// Flash attention, causal, multi-query. One wave per 16-row Q tile.
// Fixed-size strides fold into immediate offsets; per-iteration address work
// is two pointer increments. V loads issued early, hidden by softmax VALU.
// ---------------------------------------------------------------------------
__global__ __launch_bounds__(32)
void flash_attn(const _Float16* __restrict__ Qh, const _Float16* __restrict__ Kf,
                const _Float16* __restrict__ Vt, _Float16* __restrict__ Attn)
{
    __shared__ _Float16 Plds[16][40];          // row stride 80B -> 16B aligned
    constexpr int T = SEQ_T;
    constexpr int VS = 16 * T;                 // per-nt V stride (immediate)
    const int bh = blockIdx.y;
    const int b  = bh / NHEADS;
    const int h  = bh % NHEADS;
    const int t0 = blockIdx.x * 16;
    const int lane = threadIdx.x;
    const int half = lane >> 4;
    const int lr   = lane & 15;
    const int kb   = half * 16;
    const int kbB  = half * 16;

    // per-lane stepped base pointers
    const _Float16* pK = Kf + ((size_t)b * T + lr) * HDIM + kbB;
    const _Float16* pV = Vt + ((size_t)b * HDIM + lr) * T + kbB;

    // Q fragments for K-dim chunks 0..31 and 32..63
    const _Float16* qrow = Qh + ((size_t)bh * T + t0 + lr) * HDIM;
    v16h qa[2];
    #pragma unroll
    for (int c = 0; c < 2; ++c)
        qa[c] = frag2(qrow + c * 32 + kb, qrow + c * 32 + 8 + kb);

    v8f o[4] = {{}, {}, {}, {}};
    float mr[8], ls[8];
    #pragma unroll
    for (int r = 0; r < 8; ++r) { mr[r] = -3.0e38f; ls[r] = 0.0f; }

    const int s_end = t0 + 16;                 // causal horizon
    const float scale = 0.125f;                // 1/sqrt(64)

    for (int s0 = 0; s0 < s_end; s0 += 32, pK += 32 * HDIM, pV += 32) {
        // ---- scores: two 16x16 subtiles over keys [s0, s0+32)
        v8f sa[2] = {{}, {}};
        const bool act1 = (s0 + 16) < s_end;   // wave-uniform
        #pragma unroll
        for (int st = 0; st < 2; ++st) {
            if (st == 1 && !act1) continue;
            #pragma unroll
            for (int c = 0; c < 2; ++c) {
                const v16h kfr = frag2(pK + st * 16 * HDIM + c * 32,
                                       pK + st * 16 * HDIM + c * 32 + 8);
                sa[st] = __builtin_amdgcn_wmma_f32_16x16x32_f16(
                    false, qa[c], false, kfr, (short)0, sa[st], false, false);
            }
        }

        // ---- issue V fragment loads early (consumed after softmax)
        v16h vb[4];
        #pragma unroll
        for (int nt = 0; nt < 4; ++nt)
            vb[nt] = frag2(pV + nt * VS, pV + nt * VS + 8);

        // ---- prefetch next key block (stays in-bounds: rows < s_end <= T)
        if (s0 + 64 <= s_end) {
            __builtin_prefetch((const void*)(pK + 32 * HDIM), 0, 1);
            __builtin_prefetch((const void*)(pK + 48 * HDIM), 0, 1);
            #pragma unroll
            for (int nt = 0; nt < 4; ++nt)
                __builtin_prefetch((const void*)(pV + nt * VS + 32), 0, 1);
        }

        // ---- online softmax update (rows r+8*half, cols = lr per subtile)
        float p0[8], p1[8];
        #pragma unroll
        for (int r = 0; r < 8; ++r) {
            const int t = t0 + r + 8 * half;
            const float v0 = ((s0 + lr) <= t) ? sa[0][r] * scale : -3.0e38f;
            const float v1 = (act1 && (s0 + 16 + lr) <= t) ? sa[1][r] * scale
                                                           : -3.0e38f;
            const float rm = red_max16(fmaxf(v0, v1));
            const float mn = fmaxf(mr[r], rm);
            const float alpha = __expf(mr[r] - mn);
            mr[r] = mn;
            const float e0 = __expf(v0 - mn);
            const float e1 = __expf(v1 - mn);
            ls[r] = ls[r] * alpha + red_sum16(e0 + e1);
            o[0][r] *= alpha; o[1][r] *= alpha;
            o[2][r] *= alpha; o[3][r] *= alpha;
            p0[r] = e0; p1[r] = e1;
        }

        // ---- transpose P (C-layout) into A-layout via LDS
        #pragma unroll
        for (int r = 0; r < 8; ++r) {
            const int row = r + 8 * half;
            Plds[row][lr]      = (_Float16)p0[r];
            Plds[row][16 + lr] = (_Float16)p1[r];
        }
        __syncthreads();
        const _Float16* prow = &Plds[lr][0];
        const v16h pa = frag2(prow + kb, prow + 8 + kb);
        __syncthreads();

        // ---- O(16x64) += P(16x32) @ V(32x64)
        #pragma unroll
        for (int nt = 0; nt < 4; ++nt)
            o[nt] = __builtin_amdgcn_wmma_f32_16x16x32_f16(
                false, pa, false, vb[nt], (short)0, o[nt], false, false);
    }

    // ---- normalize and write attn in (B,T,C) layout, f16
    #pragma unroll
    for (int r = 0; r < 8; ++r) {
        const float inv = 1.0f / ls[r];
        const int t = t0 + r + 8 * half;
        _Float16* orow = Attn + ((size_t)(b * T + t)) * (NHEADS * HDIM) + h * HDIM;
        #pragma unroll
        for (int nt = 0; nt < 4; ++nt)
            orow[nt * 16 + lr] = (_Float16)(o[nt][r] * inv);
    }
}

// ---------------------------------------------------------------------------
extern "C" void kernel_launch(void* const* d_in, const int* in_sizes, int n_in,
                              void* d_out, int out_size, void* d_ws, size_t ws_size,
                              hipStream_t stream)
{
    const int C = DIM_C, H = NHEADS, D = HDIM, T = SEQ_T;
    const int B = in_sizes[0] / (T * C);
    const int M = B * T;

    const float* x  = (const float*)d_in[0];
    const float* Wq = (const float*)d_in[1];
    const float* bq = (const float*)d_in[2];
    const float* Wk = (const float*)d_in[3];
    const float* bk = (const float*)d_in[4];
    const float* Wv = (const float*)d_in[5];
    const float* bv = (const float*)d_in[6];
    const float* Wo = (const float*)d_in[7];
    const float* bo = (const float*)d_in[8];
    float* out = (float*)d_out;

    // workspace carve-up (256B aligned)
    char* ws = (char*)d_ws;
    size_t off = 0;
    auto carve = [&](size_t bytes) {
        char* p = ws + off;
        off += (bytes + 255) & ~(size_t)255;
        return p;
    };
    _Float16* xh   = (_Float16*)carve((size_t)M * C * 2);
    _Float16* wqt  = (_Float16*)carve((size_t)C * C * 2);
    _Float16* wkt  = (_Float16*)carve((size_t)C * D * 2);
    _Float16* wvt  = (_Float16*)carve((size_t)C * D * 2);
    _Float16* wot  = (_Float16*)carve((size_t)C * C * 2);
    float*    q32  = (float*)   carve((size_t)M * C * 4);
    float*    k32  = (float*)   carve((size_t)M * D * 4);
    float*    v32  = (float*)   carve((size_t)M * D * 4);
    _Float16* qh   = (_Float16*)carve((size_t)M * C * 2);
    _Float16* kf   = (_Float16*)carve((size_t)M * D * 2);
    _Float16* vt   = (_Float16*)carve((size_t)M * D * 2);
    _Float16* attn = (_Float16*)carve((size_t)M * C * 2);
    (void)ws_size;

    // 0) pack activations + weights into WMMA-friendly f16 layouts
    cvt_kernel<<<(M * C + 255) / 256, 256, 0, stream>>>(x, xh, M * C);
    wtrans_kernel<<<(C * C + 255) / 256, 256, 0, stream>>>(Wq, wqt, C, C);
    wtrans_kernel<<<(C * D + 255) / 256, 256, 0, stream>>>(Wk, wkt, C, D);
    wtrans_kernel<<<(C * D + 255) / 256, 256, 0, stream>>>(Wv, wvt, C, D);
    wtrans_kernel<<<(C * C + 255) / 256, 256, 0, stream>>>(Wo, wot, C, C);

    // 1) QKV projections (WMMA f16, f32 accumulate); K == C == 1024 always
    gemm_wmma<DIM_C><<<dim3(C / 64, M / 64), 128, 0, stream>>>(xh, wqt, bq, q32, C);
    gemm_wmma<DIM_C><<<dim3(1,      M / 64), 128, 0, stream>>>(xh, wkt, bk, k32, D);
    gemm_wmma<DIM_C><<<dim3(1,      M / 64), 128, 0, stream>>>(xh, wvt, bv, v32, D);

    // 2) RoPE + pack to f16; V transposed to (B,D,T)
    rope_kernel<<<(B * T * H * 32 + 255) / 256, 256, 0, stream>>>(q32, qh, B, T, H);
    rope_kernel<<<(B * T * 32 + 255) / 256, 256, 0, stream>>>(k32, kf, B, T, 1);
    vtrans_kernel<<<(M * D + 255) / 256, 256, 0, stream>>>(v32, vt, B);

    // 3) causal flash attention (multi-query: shared K/V per batch)
    flash_attn<<<dim3(T / 16, B * H), 32, 0, stream>>>(qh, kf, vt, attn);

    // 4) output projection
    gemm_wmma<DIM_C><<<dim3(C / 64, M / 64), 128, 0, stream>>>(attn, wot, bo, out, C);
}
